// Attn_40896678592626
// MI455X (gfx1250) — matrix operand order repaired and checked by reading, not compiled, not run on hardware
//
#include <hip/hip_runtime.h>
#include <stdint.h>

// MI455X / gfx1250 sigmoid-attention, WMMA-based (v_wmma_f32_16x16x32_f16)
// with double-buffered async global->LDS staging
// (global_load_async_to_lds_b128 + s_wait_asynccnt pipelining).
// B=4, S=1024, F=256 (head dim), H=8.  Sigmoid is elementwise -> streaming
// accumulation over j-tiles; the attention matrix never touches memory.

typedef __attribute__((ext_vector_type(16))) _Float16 v16h;
typedef __attribute__((ext_vector_type(8)))  _Float16 v8h;
typedef __attribute__((ext_vector_type(8)))  float    v8f;

#define WMMA_F16(a, b, c) \
  __builtin_amdgcn_wmma_f32_16x16x32_f16(false, (a), false, (b), (short)0, (c), false, false)

constexpr int Bn = 4, Sn = 1024, Fn = 256, Hn = 8;
constexpr int FH = Fn * Hn;                 // 2048
constexpr float LOGIT_SCALE = 16.0f;        // sqrt(F)

__device__ __forceinline__ float sigmoidf_dev(float x) {
  return 1.0f / (1.0f + __expf(-x));
}

// ---------------------------------------------------------------------------
// Kernel 0: weight prep.  Wq/Wk/Wv: f32 [256][2048] -> f16 [2048][256] (T).
// Wo: f32 [2048][256] -> f16 [256][2048] (T).  Makes every WMMA B-fragment a
// single contiguous 32B vector load.
// ---------------------------------------------------------------------------
__global__ __launch_bounds__(256) void prep_weights_kernel(
    const float* __restrict__ Wq, const float* __restrict__ Wk,
    const float* __restrict__ Wv, const float* __restrict__ Wo,
    _Float16* __restrict__ WqT, _Float16* __restrict__ WkT,
    _Float16* __restrict__ WvT, _Float16* __restrict__ WoT)
{
  const int idx = blockIdx.x * 256 + threadIdx.x;       // 0 .. 524287
  const int z = blockIdx.y;
  if (z < 3) {
    const float* src = (z == 0) ? Wq : (z == 1) ? Wk : Wv;
    _Float16*    dst = (z == 0) ? WqT : (z == 1) ? WkT : WvT;
    const int c = idx >> 8, k = idx & 255;              // out[c][k] = in[k][c]
    dst[idx] = (_Float16)src[(size_t)k * FH + c];
  } else {
    const int n = idx >> 11, k = idx & 2047;            // out[n][k] = in[k][n]
    WoT[idx] = (_Float16)Wo[(size_t)k * Fn + n];
  }
}

// ---------------------------------------------------------------------------
// Kernel 1: Q/K/V projection.  One wave computes one 16x16 f16 tile.
// Q,K layout: [B,H,S,F];  V layout: [B,H,F,S] (transposed -> contiguous P.V
// B-frags AND contiguous 16B stores here).
// ---------------------------------------------------------------------------
__global__ __launch_bounds__(32) void qkv_proj_kernel(
    const float* __restrict__ X,
    const _Float16* __restrict__ WqT, const _Float16* __restrict__ WkT,
    const _Float16* __restrict__ WvT,
    _Float16* __restrict__ Qh, _Float16* __restrict__ Kh, _Float16* __restrict__ Vt)
{
  const int lane = threadIdx.x & 31;
  const int m0 = blockIdx.x * 16;            // flat row in [0, B*S)
  const int h  = blockIdx.y >> 4;            // head
  const int f0 = (blockIdx.y & 15) * 16;     // feature tile within head
  const _Float16* WT = (blockIdx.z == 0) ? WqT : (blockIdx.z == 1) ? WkT : WvT;

  const int arow = m0 + (lane & 15);
  const int aoff = (lane >> 4) * 16;         // A frag K sub-offset
  const int n    = lane & 15;                // B frag column
  const int bko  = (lane >> 4) * 16;         // B frag K offset
  const int wcol = (f0 + n) * Hn + h;        // weight column f*H + h

  v8f acc = {};
  for (int k0 = 0; k0 < Fn; k0 += 32) {
    v16h a;
#pragma unroll
    for (int i = 0; i < 8; ++i) {
      a[i]     = (_Float16)X[(size_t)arow * Fn + k0 + aoff + i];
      a[8 + i] = (_Float16)X[(size_t)arow * Fn + k0 + aoff + 8 + i];
    }
    v16h b = *(const v16h*)(WT + (size_t)wcol * Fn + k0 + bko);
    acc = WMMA_F16(a, b, acc);
  }

  const int bIdx = m0 >> 10;
  const int s0   = m0 & (Sn - 1);
  const int mrow = (lane >> 4) * 8;          // C layout: M = r + 8*(lane>=16)
  if (blockIdx.z == 2) {
    v8h o;
#pragma unroll
    for (int r = 0; r < 8; ++r) o[r] = (_Float16)acc[r];
    _Float16* dst = Vt + (((size_t)bIdx * Hn + h) * Fn + f0 + n) * Sn + s0 + mrow;
    *(v8h*)dst = o;
  } else {
    _Float16* Out = (blockIdx.z == 0) ? Qh : Kh;
    _Float16* dst = Out + ((size_t)bIdx * Hn + h) * Sn * Fn;
#pragma unroll
    for (int r = 0; r < 8; ++r)
      dst[(size_t)(s0 + mrow + r) * Fn + f0 + n] = (_Float16)acc[r];
  }
}

// ---------------------------------------------------------------------------
// Kernel 2: fused sigmoid attention.  One 4-wave workgroup owns a full
// 16(i) x 256(f) output tile of one (b,h).  Per 64-j chunk:
//   - each wave computes ONE distinct 16x16 logit tile (no redundancy),
//     applies sigmoid, stores f16 into shared Plds[16][64]
//   - the 256x64 V^T slice is staged via global_load_async_to_lds_b128,
//     double-buffered: batch c+1 issued before waiting on batch c
//     (in-order ASYNCcnt => s_wait_asynccnt 16 retires the older batch)
//   - each wave does P.V for its own 64-feature slice from LDS
// ---------------------------------------------------------------------------
__global__ __launch_bounds__(128) void attn_kernel(
    const _Float16* __restrict__ Qh, const _Float16* __restrict__ Kh,
    const _Float16* __restrict__ Vt, _Float16* __restrict__ WVh)
{
  __shared__ _Float16 Plds[16 * 64];         // sigmoid(P) tile        (2 KB)
  __shared__ _Float16 Vbuf[2][256 * 64];     // V^T slice, dbl-buf    (64 KB)

  const int tid  = threadIdx.x;              // 0..127
  const int lane = tid & 31;
  const int wid  = tid >> 5;                 // wave id: jt tile & f-slice
  const int bh   = blockIdx.x;               // b*H + h
  const int i0   = blockIdx.y * 16;          // query-row tile

  const _Float16* Q = Qh + (size_t)bh * Sn * Fn;
  const _Float16* K = Kh + (size_t)bh * Sn * Fn;
  const _Float16* V = Vt + (size_t)bh * Fn * Sn;   // [F][S] transposed

  const int arow = lane & 15;
  const int aoff = (lane >> 4) * 16;
  const int n    = lane & 15;
  const int bko  = (lane >> 4) * 16;
  const int mrow = (lane >> 4) * 8;
  const int fs   = wid * 64;                 // this wave's feature slice

  const uint64_t vglob = (uint64_t)(uintptr_t)V;

  // Issue one async staging batch: V^T[0..256)[jnext..jnext+64) -> Vbuf[nb].
  // 2048 x 16B chunks / 128 threads = 16 issues per lane (per wave: +16 on
  // its ASYNCcnt).
  auto issue_batch = [&](int jnext, int nb) {
    const unsigned base = (unsigned)(uintptr_t)(&Vbuf[nb][0]);
#pragma unroll
    for (int it = 0; it < 16; ++it) {
      const int idx = it * 128 + tid;
      const int f  = idx >> 3;               // 0..255
      const int ch = idx & 7;                // 8 x 16B per 128B row
      unsigned ldsoff = base + (unsigned)(f * 128 + ch * 16);
      unsigned goff   = (unsigned)((f * Sn + jnext) * 2 + ch * 16);
      asm volatile("global_load_async_to_lds_b128 %0, %1, %2"
                   :: "v"(ldsoff), "v"(goff), "s"(vglob) : "memory");
    }
  };

  // Cache the 16x256 Q tile as 8 A-fragments (64 VGPRs).
  v16h qa[8];
#pragma unroll
  for (int kk = 0; kk < 8; ++kk) {
    const _Float16* p = Q + (size_t)(i0 + arow) * Fn + kk * 32 + aoff;
    *(v8h*)&qa[kk]       = *(const v8h*)p;
    *((v8h*)&qa[kk] + 1) = *(const v8h*)(p + 8);
  }

  v8f acc[4] = {{}, {}, {}, {}};

  issue_batch(0, 0);                         // prologue: stage first V slice

  for (int j0 = 0; j0 < Sn; j0 += 64) {
    const int buf = (j0 >> 6) & 1;

    // ---- logits: this wave's 16x16 tile at columns j0 + wid*16 ----
    {
      v8f s = {};
#pragma unroll
      for (int kk = 0; kk < 8; ++kk) {
        v16h kb = *(const v16h*)(K + (size_t)(j0 + wid * 16 + n) * Fn + kk * 32 + bko);
        s = WMMA_F16(qa[kk], kb, s);
      }
#pragma unroll
      for (int r = 0; r < 8; ++r)
        Plds[(mrow + r) * 64 + wid * 16 + n] =
            (_Float16)sigmoidf_dev(LOGIT_SCALE * s[r]);
    }

    // ---- issue next V batch, retire current one (in-order ASYNCcnt) ----
    if (j0 + 64 < Sn) {
      issue_batch(j0 + 64, buf ^ 1);
      asm volatile("s_wait_asynccnt 0x10" ::: "memory");  // older 16 done
    } else {
      asm volatile("s_wait_asynccnt 0x0" ::: "memory");
    }
    __syncthreads();                         // Plds + Vbuf[buf] visible to all

    // ---- O += P @ V: 2 j-fragments x 4 f-tiles from LDS ----
#pragma unroll
    for (int jc = 0; jc < 2; ++jc) {
      v16h pa;
      const _Float16* pp = &Plds[arow * 64 + jc * 32 + aoff];
      *(v8h*)&pa       = *(const v8h*)pp;
      *((v8h*)&pa + 1) = *(const v8h*)(pp + 8);
#pragma unroll
      for (int nt = 0; nt < 4; ++nt) {
        v16h vb = *(const v16h*)(&Vbuf[buf][(fs + nt * 16 + n) * 64 + jc * 32 + bko]);
        acc[nt] = WMMA_F16(pa, vb, acc[nt]);
      }
    }
    __syncthreads();                         // done reading before next writes
  }

  // Store attention output in [b, s, f*H + h] layout for the Wo GEMM.
  const int bIdx = bh >> 3, h = bh & 7;
#pragma unroll
  for (int nt = 0; nt < 4; ++nt)
#pragma unroll
    for (int r = 0; r < 8; ++r)
      WVh[((size_t)bIdx * Sn + i0 + mrow + r) * FH + (fs + nt * 16 + n) * Hn + h] =
          (_Float16)acc[nt][r];
}

// ---------------------------------------------------------------------------
// Kernel 3: output projection + ReLU.  [4096,2048](f16) x WoT -> fp32.
// ---------------------------------------------------------------------------
__global__ __launch_bounds__(32) void out_proj_kernel(
    const _Float16* __restrict__ WVh, const _Float16* __restrict__ WoT,
    float* __restrict__ out)
{
  const int lane = threadIdx.x & 31;
  const int m0 = blockIdx.x * 16;
  const int n0 = blockIdx.y * 16;
  const int arow = lane & 15, aoff = (lane >> 4) * 16;
  const int n = lane & 15,   bko  = (lane >> 4) * 16;

  v8f acc = {};
  for (int k0 = 0; k0 < FH; k0 += 32) {
    v16h a;
    const _Float16* p = WVh + (size_t)(m0 + arow) * FH + k0 + aoff;
    *(v8h*)&a       = *(const v8h*)p;
    *((v8h*)&a + 1) = *(const v8h*)(p + 8);
    v16h b = *(const v16h*)(WoT + (size_t)(n0 + n) * FH + k0 + bko);
    acc = WMMA_F16(a, b, acc);
  }
  const int mrow = (lane >> 4) * 8;
#pragma unroll
  for (int r = 0; r < 8; ++r) {
    float v = acc[r];
    out[(size_t)(m0 + mrow + r) * Fn + n0 + n] = v > 0.0f ? v : 0.0f;
  }
}

// ---------------------------------------------------------------------------
extern "C" void kernel_launch(void* const* d_in, const int* in_sizes, int n_in,
                              void* d_out, int out_size, void* d_ws, size_t ws_size,
                              hipStream_t stream)
{
  (void)in_sizes; (void)n_in; (void)out_size; (void)ws_size;

  const float* X  = (const float*)d_in[0];
  const float* Wq = (const float*)d_in[1];
  const float* Wk = (const float*)d_in[2];
  const float* Wv = (const float*)d_in[3];
  const float* Wo = (const float*)d_in[4];
  float* out = (float*)d_out;

  const size_t elems = (size_t)Bn * Hn * Sn * Fn;   // 8,388,608 per tensor
  const size_t welem = (size_t)Fn * FH;             //   524,288 per weight
  _Float16* Qh  = (_Float16*)d_ws;                  // [B,H,S,F] f16
  _Float16* Kh  = Qh + elems;                       // [B,H,S,F] f16
  _Float16* Vt  = Kh + elems;                       // [B,H,F,S] f16 (T)
  _Float16* WVh = Vt + elems;                       // [B,S,F*H]  f16
  _Float16* WqT = WVh + elems;                      // [2048][256] f16
  _Float16* WkT = WqT + welem;
  _Float16* WvT = WkT + welem;
  _Float16* WoT = WvT + welem;                      // [256][2048] f16

  prep_weights_kernel<<<dim3(welem / 256, 4), dim3(256), 0, stream>>>(
      Wq, Wk, Wv, Wo, WqT, WkT, WvT, WoT);
  qkv_proj_kernel<<<dim3(Bn * Sn / 16, (Fn / 16) * Hn, 3), dim3(32), 0, stream>>>(
      X, WqT, WkT, WvT, Qh, Kh, Vt);
  attn_kernel<<<dim3(Bn * Hn, Sn / 16), dim3(128), 0, stream>>>(
      Qh, Kh, Vt, WVh);
  out_proj_kernel<<<dim3(Bn * Sn / 16, Fn / 16), dim3(32), 0, stream>>>(
      WVh, WoT, out);
}
